// IoUMetric_18769007083843
// MI455X (gfx1250) — 8-bit matrix operand order repaired and checked by reading, compiled, not run on hardware
//
#include <hip/hip_runtime.h>


typedef __attribute__((ext_vector_type(8))) int   v8i;
typedef __attribute__((ext_vector_type(4))) float f4;

#define CLS 21
#define HW   (512 * 512)
#define HW4  (HW / 4)

__device__ __forceinline__ unsigned onehot4(unsigned w, unsigned n) {
  unsigned x = w ^ (n * 0x01010101u);
  unsigned t = x | (x >> 1);
  t |= (t >> 2);
  t |= (x >> 4);
  return ~t & 0x01010101u;
}

__device__ __forceinline__ unsigned argmax4pack(const f4* __restrict__ p) {
  f4 v = __builtin_nontemporal_load(p);
  float b0 = v.x, b1 = v.y, b2 = v.z, b3 = v.w;
  unsigned i0 = 0, i1 = 0, i2 = 0, i3 = 0;
#pragma unroll
  for (unsigned c = 1; c < CLS; ++c) {
    f4 u = __builtin_nontemporal_load(p + (size_t)c * HW4);
    if (u.x > b0) { b0 = u.x; i0 = c; }
    if (u.y > b1) { b1 = u.y; i1 = c; }
    if (u.z > b2) { b2 = u.z; i2 = c; }
    if (u.w > b3) { b3 = u.w; i3 = c; }
  }
  return i0 | (i1 << 8) | (i2 << 16) | (i3 << 24);
}

__global__ void zero_counts_kernel(unsigned* __restrict__ g) {
  if (threadIdx.x < 64) g[threadIdx.x] = 0u;
}

__global__ void __launch_bounds__(256)
iou_hist_kernel(const float* __restrict__ outp,
                const float* __restrict__ tgtp,
                unsigned* __restrict__ gcnt) {
  __shared__ unsigned s_t[8][32];
  __shared__ unsigned s_p[8][32];
  __shared__ unsigned s_m[8][32];
  __shared__ unsigned s_hist[64];

  const unsigned tid  = threadIdx.x;
  const unsigned lane = tid & 31u;
  const unsigned wv   = tid >> 5;

  const size_t   g   = (size_t)blockIdx.x * 256u + tid;
  const size_t   P0  = g * 4u;
  const size_t   img = P0 >> 18;
  const unsigned hw  = (unsigned)(P0 & (HW - 1));

  const f4* ob = (const f4*)outp + img * (size_t)(CLS * HW4) + (hw >> 2);
  const f4* tb = (const f4*)tgtp + img * (size_t)(CLS * HW4) + (hw >> 2);

  const unsigned pw = argmax4pack(ob);
  const unsigned tw = argmax4pack(tb);
  unsigned mx = pw ^ tw;
  unsigned mt = mx | (mx >> 1); mt |= (mt >> 2); mt |= (mx >> 4);
  const unsigned mw = ~mt & 0x01010101u;

  if (tid < 64) s_hist[tid] = 0u;
  s_t[wv][lane] = tw;
  s_p[wv][lane] = pw;
  s_m[wv][lane] = mw;
  __syncthreads();

  v8i acc_t_lo = {}, acc_t_hi = {}, acc_p_lo = {}, acc_p_hi = {};

  const unsigned n    = lane & 15u;
  const unsigned hi   = lane >> 4;
  const unsigned ones = 0x01010101u;

#pragma unroll
  for (unsigned chunk = 0; chunk < 2; ++chunk) {
    const unsigned cb = chunk * 16u;

    unsigned twd[8], pwd[8];
#pragma unroll
    for (unsigned j = 0; j < 8; ++j) {
      const unsigned widx = cb + hi * 4u + (j < 4u ? j : j + 4u);
      twd[j] = s_t[wv][widx];
      pwd[j] = s_p[wv][widx];
    }
    v8i bt_lo, bt_hi, bp_lo, bp_hi;
#pragma unroll
    for (unsigned j = 0; j < 8; ++j) {
      bt_lo[j] = (int)onehot4(twd[j], n);
      bt_hi[j] = (int)onehot4(twd[j], n + 16u);
      bp_lo[j] = (int)onehot4(pwd[j], n);
      bp_hi[j] = (int)onehot4(pwd[j], n + 16u);
    }

    v8i amat;
#pragma unroll
    for (unsigned j = 0; j < 8; ++j) {
      const unsigned widx = cb + hi * 4u + (j < 4u ? j : j + 4u);
      const unsigned mword = s_m[wv][widx];
      amat[j] = (n == 0u) ? (int)ones : ((n == 1u) ? (int)mword : 0);
    }

    acc_t_lo = __builtin_amdgcn_wmma_i32_16x16x64_iu8(false, amat, false, bt_lo, acc_t_lo, false, false);
    acc_t_hi = __builtin_amdgcn_wmma_i32_16x16x64_iu8(false, amat, false, bt_hi, acc_t_hi, false, false);
    acc_p_lo = __builtin_amdgcn_wmma_i32_16x16x64_iu8(false, amat, false, bp_lo, acc_p_lo, false, false);
    acc_p_hi = __builtin_amdgcn_wmma_i32_16x16x64_iu8(false, amat, false, bp_hi, acc_p_hi, false, false);
  }

  if (lane < 16u) {
    atomicAdd(&s_hist[lane],       (unsigned)acc_t_lo[1]);
    atomicAdd(&s_hist[21u + lane], (unsigned)acc_t_lo[0]);
    atomicAdd(&s_hist[42u + lane], (unsigned)acc_p_lo[0]);
    if (lane < 5u) {
      const unsigned ch = 16u + lane;
      atomicAdd(&s_hist[ch],       (unsigned)acc_t_hi[1]);
      atomicAdd(&s_hist[21u + ch], (unsigned)acc_t_hi[0]);
      atomicAdd(&s_hist[42u + ch], (unsigned)acc_p_hi[0]);
    }
  }
  __syncthreads();

  if (tid < 63u) {
    const unsigned v = s_hist[tid];
    if (v) atomicAdd(&gcnt[tid], v);
  }
}

__global__ void finalize_kernel(const unsigned* __restrict__ cnt,
                                float* __restrict__ res) {
  if (threadIdx.x == 0) {
    float acc = 0.0f;
    for (int c = 0; c < CLS; ++c) {
      const float tp = (float)cnt[c];
      const float tg = (float)cnt[21 + c];
      const float oh = (float)cnt[42 + c];
      const float den = tg + oh - tp;
      acc += (den > 0.0f) ? (tp / den) : 0.0f;
    }
    res[0] = acc / (float)CLS;
  }
}

extern "C" void kernel_launch(void* const* d_in, const int* in_sizes, int n_in,
                              void* d_out, int out_size, void* d_ws, size_t ws_size,
                              hipStream_t stream) {
  const float* outp = (const float*)d_in[0];
  const float* tgtp = (const float*)d_in[1];
  unsigned*    cnt  = (unsigned*)d_ws;
  float*       res  = (float*)d_out;

  const long long total = (long long)in_sizes[0];
  const long long npix  = total / CLS;
  const int nthreads    = (int)(npix / 4);
  const int blocks      = (nthreads + 255) / 256;

  hipLaunchKernelGGL(zero_counts_kernel, dim3(1), dim3(64), 0, stream, cnt);
  hipLaunchKernelGGL(iou_hist_kernel, dim3(blocks), dim3(256), 0, stream,
                     outp, tgtp, cnt);
  hipLaunchKernelGGL(finalize_kernel, dim3(1), dim3(32), 0, stream, cnt, res);
}
